// MultimodalMatcher_83434034692460
// MI455X (gfx1250) — matrix operand order repaired and checked by reading, not compiled, not run on hardware
//
#include <hip/hip_runtime.h>

// ---------------------------------------------------------------------------
// Shapes
// ---------------------------------------------------------------------------
constexpr int BB = 256;   // batch
constexpr int LT = 64;    // text length
constexpr int LI = 197;   // image length
constexpr int HD = 768;   // hidden

constexpr long long BIGSZ   = (long long)BB * LI * HD;   // 38,731,776
constexpr long long SMALLSZ = (long long)BB * LT * HD;   // 12,582,912
constexpr long long PBSZ    = (long long)BB * LT * LI;   //  3,227,648
constexpr long long VECSZ   = (long long)BB * HD;        //    196,608

#if defined(__has_builtin)
#  if __has_builtin(__builtin_amdgcn_sched_barrier)
#    define SCHED_FENCE() __builtin_amdgcn_sched_barrier(0)
#  endif
#endif
#ifndef SCHED_FENCE
#  define SCHED_FENCE()
#endif

// ---------------------------------------------------------------------------
// WMMA types (probe-verified signatures for gfx1250)
// ---------------------------------------------------------------------------
typedef __attribute__((ext_vector_type(16))) __bf16     v16bf;
typedef __attribute__((ext_vector_type(8)))  float      v8f;
typedef __attribute__((ext_vector_type(4)))  unsigned   uv4;

union FragBF { v16bf v; uv4 q[2]; };

__device__ __forceinline__ unsigned short f2bf_bits(float f) {
  union { float f; unsigned u; } a; a.f = f;
  unsigned u = a.u;
  u += 0x7FFFu + ((u >> 16) & 1u);          // round-to-nearest-even
  return (unsigned short)(u >> 16);
}
__device__ __forceinline__ unsigned pk_bf(float lo, float hi) {
  return (unsigned)f2bf_bits(lo) | ((unsigned)f2bf_bits(hi) << 16);
}

// ---------------------------------------------------------------------------
// Tiled bf16 WMMA GEMM.
//   BNK = true :  C[b] = A[b](MxK) * Bm[b](NxK)^T   (Linear / cosine-sim)
//   BNK = false:  C[b] = A[b](MxK) * Bm[b](KxN)     (transport einsum)
// f32 in HBM; bf16 staged in LDS; f32 accumulate via v_wmma_f32_16x16x32_bf16.
// Block: 128 threads = 4 waves; block tile 64(M) x 64(N); wave tile 16x64.
// Interior tiles take a vectorized staging path (b128 loads + ds_store_b64);
// edge tiles fall back to bounds-checked scalar staging.
// A sched_barrier separates the fragment ds_loads from the WMMA chain so all
// ten ds_load_b128s are in flight behind a single s_wait_dscnt, and the four
// WMMAs issue back-to-back into the XDL pipe (independent accumulators).
// ---------------------------------------------------------------------------
template <bool BNK>
__global__ __launch_bounds__(128)
void gemm_bf16_wmma(const float* __restrict__ A, const float* __restrict__ Bm,
                    const float* __restrict__ bias, float* __restrict__ C,
                    int M, int N, int K,
                    long long sA, long long sB, long long sC)
{
  // 80-byte row stride: 16B-aligned b128 fragment loads, no bank conflicts
  __shared__ alignas(16) unsigned short lA[64][40];
  __shared__ alignas(16) unsigned short lB[64][40];

  const int b    = blockIdx.z;
  const float* Ab = A  + (long long)b * sA;
  const float* Bb = Bm + (long long)b * sB;
  float*       Cb = C  + (long long)b * sC;

  const int n0   = blockIdx.x * 64;
  const int m0   = blockIdx.y * 64;
  const int tid  = threadIdx.x;
  const int wave = tid >> 5;
  const int lane = tid & 31;

  const bool aFullM = (m0 + 64 <= M) && ((K & 3) == 0);
  const bool bFullN = (n0 + 64 <= N);

  v8f acc[4] = {};

  for (int k0 = 0; k0 < K; k0 += 32) {
    const bool fullK = (k0 + 32 <= K);
    __syncthreads();

    // ---- stage A tile 64x32 -> lA[r][k] ----
    if (aFullM && fullK) {
      #pragma unroll
      for (int it = 0; it < 4; ++it) {
        int q  = tid + it * 128;          // 0..511 float4 quads
        int r  = q >> 3;                  // row 0..63
        int c4 = (q & 7) << 2;            // k 0,4,...,28
        const float4 v = *(const float4*)(Ab + (long long)(m0 + r) * K + (k0 + c4));
        uint2 p; p.x = pk_bf(v.x, v.y); p.y = pk_bf(v.z, v.w);
        *(uint2*)&lA[r][c4] = p;
      }
    } else {
      #pragma unroll
      for (int it = 0; it < 16; ++it) {
        int idx = tid + it * 128;
        int r = idx >> 5, c = idx & 31;
        int gm = m0 + r, gk = k0 + c;
        float v = (gm < M && gk < K) ? Ab[(long long)gm * K + gk] : 0.f;
        lA[r][c] = f2bf_bits(v);
      }
    }

    // ---- stage B tile -> lB[n][k] ----
    if (BNK) {
      if (bFullN && fullK && ((K & 3) == 0)) {
        #pragma unroll
        for (int it = 0; it < 4; ++it) {
          int q  = tid + it * 128;
          int r  = q >> 3;
          int c4 = (q & 7) << 2;
          const float4 v = *(const float4*)(Bb + (long long)(n0 + r) * K + (k0 + c4));
          uint2 p; p.x = pk_bf(v.x, v.y); p.y = pk_bf(v.z, v.w);
          *(uint2*)&lB[r][c4] = p;
        }
      } else {
        #pragma unroll
        for (int it = 0; it < 16; ++it) {
          int idx = tid + it * 128;
          int r = idx >> 5, c = idx & 31;
          int gn = n0 + r, gk = k0 + c;
          float v = (gn < N && gk < K) ? Bb[(long long)gn * K + gk] : 0.f;
          lB[r][c] = f2bf_bits(v);
        }
      }
    } else {
      if (bFullN && fullK) {
        #pragma unroll
        for (int it = 0; it < 4; ++it) {
          int q  = tid + it * 128;
          int n  = q & 63;                  // coalesced across lanes
          int kk = (q >> 6) << 2;           // k 0,4,...,28
          const float* src = Bb + (long long)(k0 + kk) * N + (n0 + n);
          float a0 = src[0];
          float a1 = src[(long long)N];
          float a2 = src[(long long)2 * N];
          float a3 = src[(long long)3 * N];
          uint2 p; p.x = pk_bf(a0, a1); p.y = pk_bf(a2, a3);
          *(uint2*)&lB[n][kk] = p;
        }
      } else {
        #pragma unroll
        for (int it = 0; it < 8; ++it) {
          int idx = tid + it * 128;         // 0..1023
          int n = idx & 63, kk = idx >> 6;  // kk 0..15
          int gn = n0 + n;
          int gk = k0 + kk;
          float v = (gn < N && gk < K) ? Bb[(long long)gk * N + gn] : 0.f;
          lB[n][kk] = f2bf_bits(v);
          int gk2 = k0 + kk + 16;
          float v2 = (gn < N && gk2 < K) ? Bb[(long long)gk2 * N + gn] : 0.f;
          lB[n][kk + 16] = f2bf_bits(v2);
        }
      }
    }
    __syncthreads();

    // ---- fragments (wave32 layouts per CDNA5 ISA 7.12.2) ----
    const int mrow  = lane & 15;
    const int khalf = lane >> 4;
    // A 16x32: lanes<16 hold K 0..7 & 16..23, lanes>=16 hold K 8..15 & 24..31
    FragBF fa;
    fa.q[0] = *(const uv4*)&lA[16 * wave + mrow][8 * khalf];
    fa.q[1] = *(const uv4*)&lA[16 * wave + mrow][16 + 8 * khalf];
    // B 32x16: lanes<16 hold K 0..7 & 16..23, lanes>=16 hold K 8..15 & 24..31
    FragBF fb0, fb1, fb2, fb3;
    fb0.q[0] = *(const uv4*)&lB[ 0 + mrow][8 * khalf];
    fb0.q[1] = *(const uv4*)&lB[ 0 + mrow][16 + 8 * khalf];
    fb1.q[0] = *(const uv4*)&lB[16 + mrow][8 * khalf];
    fb1.q[1] = *(const uv4*)&lB[16 + mrow][16 + 8 * khalf];
    fb2.q[0] = *(const uv4*)&lB[32 + mrow][8 * khalf];
    fb2.q[1] = *(const uv4*)&lB[32 + mrow][16 + 8 * khalf];
    fb3.q[0] = *(const uv4*)&lB[48 + mrow][8 * khalf];
    fb3.q[1] = *(const uv4*)&lB[48 + mrow][16 + 8 * khalf];
    // Keep all fragment loads above, all WMMAs below: one dscnt wait, then
    // four back-to-back matrix ops.
    SCHED_FENCE();
    acc[0] = __builtin_amdgcn_wmma_f32_16x16x32_bf16(
               false, fa.v, false, fb0.v, (short)0, acc[0], false, false);
    acc[1] = __builtin_amdgcn_wmma_f32_16x16x32_bf16(
               false, fa.v, false, fb1.v, (short)0, acc[1], false, false);
    acc[2] = __builtin_amdgcn_wmma_f32_16x16x32_bf16(
               false, fa.v, false, fb2.v, (short)0, acc[2], false, false);
    acc[3] = __builtin_amdgcn_wmma_f32_16x16x32_bf16(
               false, fa.v, false, fb3.v, (short)0, acc[3], false, false);
  }

  // ---- store D (32-bit C/D layout: vgpr r -> M = r + 8*(lane>=16)) ----
  const int nlane = lane & 15;
  const int mhalf = lane >> 4;
  #pragma unroll
  for (int t = 0; t < 4; ++t) {
    int gn = n0 + 16 * t + nlane;
    if (gn >= N) continue;
    float bv = bias ? bias[gn] : 0.f;
    #pragma unroll
    for (int r = 0; r < 8; ++r) {
      int gm = m0 + 16 * wave + 8 * mhalf + r;
      if (gm < M) Cb[(long long)gm * N + gn] = acc[t][r] + bv;
    }
  }
}

// ---------------------------------------------------------------------------
// Row L2 normalize (H = 768, one wave32 per row)
// ---------------------------------------------------------------------------
__global__ __launch_bounds__(256)
void l2norm_rows(const float* __restrict__ in, float* __restrict__ out, int rows)
{
  int row  = blockIdx.x * 8 + (threadIdx.x >> 5);
  int lane = threadIdx.x & 31;
  if (row >= rows) return;
  const float* xr = in + (long long)row * HD;
  float v[24]; float s = 0.f;
  #pragma unroll
  for (int i = 0; i < 24; ++i) { v[i] = xr[lane + 32 * i]; s += v[i] * v[i]; }
  #pragma unroll
  for (int m = 16; m; m >>= 1) s += __shfl_xor(s, m, 32);
  float inv = 1.f / fmaxf(sqrtf(s), 1e-12f);
  float* orow = out + (long long)row * HD;
  #pragma unroll
  for (int i = 0; i < 24; ++i) orow[lane + 32 * i] = v[i] * inv;
}

// ---------------------------------------------------------------------------
// out-row = LayerNorm(x + res) (in-place on x), H = 768, wave per row
// ---------------------------------------------------------------------------
__global__ __launch_bounds__(256)
void add_ln_rows(float* __restrict__ x, const float* __restrict__ res,
                 const float* __restrict__ g, const float* __restrict__ be, int rows)
{
  int row  = blockIdx.x * 8 + (threadIdx.x >> 5);
  int lane = threadIdx.x & 31;
  if (row >= rows) return;
  float* xr = x + (long long)row * HD;
  const float* rr = res + (long long)row * HD;
  float v[24]; float s = 0.f;
  #pragma unroll
  for (int i = 0; i < 24; ++i) { v[i] = xr[lane + 32 * i] + rr[lane + 32 * i]; s += v[i]; }
  #pragma unroll
  for (int m = 16; m; m >>= 1) s += __shfl_xor(s, m, 32);
  float mean = s * (1.f / 768.f);
  float vs = 0.f;
  #pragma unroll
  for (int i = 0; i < 24; ++i) { float d = v[i] - mean; vs += d * d; }
  #pragma unroll
  for (int m = 16; m; m >>= 1) vs += __shfl_xor(vs, m, 32);
  float rstd = rsqrtf(vs * (1.f / 768.f) + 1e-5f);
  #pragma unroll
  for (int i = 0; i < 24; ++i) {
    int h = lane + 32 * i;
    xr[h] = (v[i] - mean) * rstd * g[h] + be[h];
  }
}

// ---------------------------------------------------------------------------
// Sinkhorn: K=exp(C/eps); 10x (row-norm over last dim, col-norm over dim -2);
// then L2-normalize rows. Entire RxCc matrix lives in LDS (<= 64x197 f32).
// One workgroup per batch element.
// ---------------------------------------------------------------------------
__global__ __launch_bounds__(256)
void sinkhorn_kernel(float* __restrict__ P, int R, int Cc)
{
  __shared__ float sm[LT * LI];          // 12608 floats, max of both orders
  __shared__ float rsum[208];
  __shared__ float csum[208];
  const int b = blockIdx.x, tid = threadIdx.x;
  const int n = R * Cc;
  float* Pb = P + (long long)b * n;

  for (int i = tid; i < n; i += 256) sm[i] = expf(Pb[i] * 10.f);  // 1/eps = 10
  __syncthreads();

  for (int it = 0; it < 10; ++it) {
    for (int r = tid; r < R; r += 256) {
      float s = 0.f;
      for (int c = 0; c < Cc; ++c) s += sm[r * Cc + c];
      rsum[r] = 1.f / s;
    }
    __syncthreads();
    for (int i = tid; i < n; i += 256) sm[i] *= rsum[i / Cc];
    __syncthreads();
    for (int c = tid; c < Cc; c += 256) {
      float s = 0.f;
      for (int r = 0; r < R; ++r) s += sm[r * Cc + c];
      csum[c] = 1.f / s;
    }
    __syncthreads();
    for (int i = tid; i < n; i += 256) sm[i] *= csum[i % Cc];
    __syncthreads();
  }
  // final L2 row normalize
  for (int r = tid; r < R; r += 256) {
    float s = 0.f;
    for (int c = 0; c < Cc; ++c) { float v = sm[r * Cc + c]; s += v * v; }
    rsum[r] = 1.f / fmaxf(sqrtf(s), 1e-12f);
  }
  __syncthreads();
  for (int i = tid; i < n; i += 256) Pb[i] = sm[i] * rsum[i / Cc];
}

// ---------------------------------------------------------------------------
// Soft pool over implicit concat [x1 (L1 rows); x2 (L2 rows)], then LayerNorm.
// One workgroup per batch element. Logits: one wave32 per row (coalesced).
// ---------------------------------------------------------------------------
__global__ __launch_bounds__(256)
void softpool_kernel(const float* __restrict__ x1, int L1,
                     const float* __restrict__ x2, int L2,
                     const float* __restrict__ fw, const float* __restrict__ fb,
                     const float* __restrict__ g,  const float* __restrict__ be,
                     float* __restrict__ out)
{
  __shared__ float lg[512];
  __shared__ float red[256];
  __shared__ float sv[HD];
  const int b = blockIdx.x, tid = threadIdx.x;
  const int wv = tid >> 5, lane = tid & 31;
  const float* X1 = x1 + (long long)b * L1 * HD;
  const float* X2 = x2 + (long long)b * L2 * HD;
  const int L = L1 + L2;

  // logits: wave per row, lanes stride the 768 columns (coalesced)
  for (int l = wv; l < L; l += 8) {
    const float* r = (l < L1) ? (X1 + (long long)l * HD) : (X2 + (long long)(l - L1) * HD);
    float s = 0.f;
    #pragma unroll
    for (int i = 0; i < 24; ++i) { int h = lane + 32 * i; s += r[h] * fw[h]; }
    #pragma unroll
    for (int m = 16; m; m >>= 1) s += __shfl_xor(s, m, 32);
    if (lane == 0) lg[l] = s + fb[0];
  }
  __syncthreads();
  // softmax over L
  float m = -1e30f;
  for (int l = tid; l < L; l += 256) m = fmaxf(m, lg[l]);
  red[tid] = m; __syncthreads();
  for (int st = 128; st; st >>= 1) { if (tid < st) red[tid] = fmaxf(red[tid], red[tid + st]); __syncthreads(); }
  m = red[0]; __syncthreads();
  float s = 0.f;
  for (int l = tid; l < L; l += 256) { float e = expf(lg[l] - m); lg[l] = e; s += e; }
  red[tid] = s; __syncthreads();
  for (int st = 128; st; st >>= 1) { if (tid < st) red[tid] += red[tid + st]; __syncthreads(); }
  float inv = 1.f / red[0];
  __syncthreads();
  // weighted sum over rows (h coalesced across threads)
  for (int h = tid; h < HD; h += 256) {
    float acc = 0.f;
    for (int l = 0; l < L1; ++l) acc += X1[(long long)l * HD + h] * lg[l];
    for (int l = 0; l < L2; ++l) acc += X2[(long long)l * HD + h] * lg[L1 + l];
    sv[h] = acc * inv;
  }
  __syncthreads();
  // LayerNorm over H
  float ls = 0.f;
  for (int h = tid; h < HD; h += 256) ls += sv[h];
  red[tid] = ls; __syncthreads();
  for (int st = 128; st; st >>= 1) { if (tid < st) red[tid] += red[tid + st]; __syncthreads(); }
  float mean = red[0] * (1.f / 768.f); __syncthreads();
  float lv = 0.f;
  for (int h = tid; h < HD; h += 256) { float d = sv[h] - mean; lv += d * d; }
  red[tid] = lv; __syncthreads();
  for (int st = 128; st; st >>= 1) { if (tid < st) red[tid] += red[tid + st]; __syncthreads(); }
  float rstd = rsqrtf(red[0] * (1.f / 768.f) + 1e-5f);
  for (int h = tid; h < HD; h += 256)
    out[(long long)b * HD + h] = (sv[h] - mean) * rstd * g[h] + be[h];
}

// ---------------------------------------------------------------------------
// score[i][j] = 0.5 * (men0[i].ent0[j] + men1[i].ent1[j])
// ---------------------------------------------------------------------------
__global__ __launch_bounds__(256)
void score_kernel(const float* __restrict__ m0, const float* __restrict__ e0,
                  const float* __restrict__ m1, const float* __restrict__ e1,
                  float* __restrict__ out)
{
  int j = blockIdx.x * 16 + threadIdx.x;   // entity
  int i = blockIdx.y * 16 + threadIdx.y;   // mention
  const float* a0 = m0 + (long long)i * HD;
  const float* b0 = e0 + (long long)j * HD;
  const float* a1 = m1 + (long long)i * HD;
  const float* b1 = e1 + (long long)j * HD;
  float s = 0.f;
  for (int k = 0; k < HD; ++k) s += a0[k] * b0[k] + a1[k] * b1[k];
  out[(long long)i * BB + j] = 0.5f * s;
}

// ---------------------------------------------------------------------------
// Host-side orchestration
// ---------------------------------------------------------------------------
struct OT {
  const float *ori_w, *ori_b, *tar_w, *tar_b, *ftgt_w, *ftgt_b;
  const float *fo_w, *fo_b, *fot_w, *fot_b, *ln_g, *ln_b;
};

static inline void launch_gemm(hipStream_t st, const float* A, const float* Bm,
                               const float* bias, float* C,
                               int M, int N, int K, int batch,
                               long long sA, long long sB, long long sC, bool bnk)
{
  dim3 grid((N + 63) / 64, (M + 63) / 64, batch);
  if (bnk) gemm_bf16_wmma<true ><<<grid, 128, 0, st>>>(A, Bm, bias, C, M, N, K, sA, sB, sC);
  else     gemm_bf16_wmma<false><<<grid, 128, 0, st>>>(A, Bm, bias, C, M, N, K, sA, sB, sC);
}

static inline void launch_l2(hipStream_t st, const float* in, float* out, int rows) {
  l2norm_rows<<<(rows + 7) / 8, 256, 0, st>>>(in, out, rows);
}
static inline void launch_addln(hipStream_t st, float* x, const float* res,
                                const float* g, const float* b, int rows) {
  add_ln_rows<<<(rows + 7) / 8, 256, 0, st>>>(x, res, g, b, rows);
}

static void run_side(hipStream_t st, const float* text, const float* img,
                     const OT& i2t, const OT& t2i,
                     const float* ftW, const float* ftB,
                     const float* fiW, const float* fiB,
                     const float* p0w, const float* p0b, const float* p0g, const float* p0be,
                     const float* p1w, const float* p1b, const float* p1g, const float* p1be,
                     float* big1, float* big2, float* big3,
                     float* s1, float* s2, float* s3, float* s4, float* pb,
                     float* out0, float* out1)
{
  const int MT = BB * LT, MI = BB * LI;
  // ---- i2t transport (origin = image, target = text) -> s4 [B,LT,H]
  launch_gemm(st, img,  i2t.ori_w, i2t.ori_b,  big1, MI, HD, HD, 1, 0, 0, 0, true);  // o
  launch_gemm(st, text, i2t.tar_w, i2t.tar_b,  s1,   MT, HD, HD, 1, 0, 0, 0, true);  // t0
  launch_gemm(st, s1,   i2t.ftgt_w, i2t.ftgt_b, s2,  MT, HD, HD, 1, 0, 0, 0, true);  // t
  launch_gemm(st, big1, i2t.fo_w,  i2t.fo_b,   big2, MI, HD, HD, 1, 0, 0, 0, true);  // o_fc
  launch_gemm(st, big1, i2t.fot_w, i2t.fot_b,  big3, MI, HD, HD, 1, 0, 0, 0, true);  // o_tr
  launch_l2(st, s2, s3, MT);
  launch_l2(st, big2, big2, MI);
  launch_gemm(st, s3, big2, nullptr, pb, LT, LI, HD, BB,
              (long long)LT * HD, (long long)LI * HD, (long long)LT * LI, true);      // cosine
  sinkhorn_kernel<<<BB, 256, 0, st>>>(pb, LT, LI);
  launch_gemm(st, pb, big3, nullptr, s4, LT, HD, LI, BB,
              (long long)LT * LI, (long long)LI * HD, (long long)LT * HD, false);     // transport
  launch_addln(st, s4, s2, i2t.ln_g, i2t.ln_b, MT);

  // ---- t2i transport (origin = text, target = image) -> big1 [B,LI,H]
  launch_gemm(st, text, t2i.ori_w, t2i.ori_b,  s1,   MT, HD, HD, 1, 0, 0, 0, true);
  launch_gemm(st, img,  t2i.tar_w, t2i.tar_b,  big1, MI, HD, HD, 1, 0, 0, 0, true);
  launch_gemm(st, big1, t2i.ftgt_w, t2i.ftgt_b, big2, MI, HD, HD, 1, 0, 0, 0, true);
  launch_gemm(st, s1,   t2i.fo_w,  t2i.fo_b,   s2,   MT, HD, HD, 1, 0, 0, 0, true);
  launch_gemm(st, s1,   t2i.fot_w, t2i.fot_b,  s3,   MT, HD, HD, 1, 0, 0, 0, true);
  launch_l2(st, big2, big3, MI);
  launch_l2(st, s2, s2, MT);
  launch_gemm(st, big3, s2, nullptr, pb, LI, LT, HD, BB,
              (long long)LI * HD, (long long)LT * HD, (long long)LI * LT, true);
  sinkhorn_kernel<<<BB, 256, 0, st>>>(pb, LI, LT);
  launch_gemm(st, pb, s3, nullptr, big1, LI, HD, LT, BB,
              (long long)LI * LT, (long long)LT * HD, (long long)LI * HD, false);
  launch_addln(st, big1, big2, t2i.ln_g, t2i.ln_b, MI);

  // ---- projections
  launch_gemm(st, text, ftW, ftB, s1,   MT, HD, HD, 1, 0, 0, 0, true);
  launch_gemm(st, img,  fiW, fiB, big2, MI, HD, HD, 1, 0, 0, 0, true);

  // ---- soft pools over implicit concats
  softpool_kernel<<<BB, 256, 0, st>>>(s1,   LT, s4,   LT, p0w, p0b, p0g, p0be, out0);
  softpool_kernel<<<BB, 256, 0, st>>>(big1, LI, big2, LI, p1w, p1b, p1g, p1be, out1);
}

extern "C" void kernel_launch(void* const* d_in, const int* in_sizes, int n_in,
                              void* d_out, int out_size, void* d_ws, size_t ws_size,
                              hipStream_t stream)
{
  (void)in_sizes; (void)n_in; (void)out_size; (void)ws_size;
  auto in = [&](int i) { return (const float*)d_in[i]; };

  // JAX pytree flatten order: dict keys sorted alphabetically at every level.
  // params: fc_image(W,b) fc_text(W,b)
  //         i2t{fc_origin, fc_origin_transport, fc_target, ln, trans_ori, trans_tar}
  //         sp0{fc, ln} sp1{fc, ln} t2i{...}
  const float *fiW = in(0),  *fiB = in(1);
  const float *ftW = in(2),  *ftB = in(3);
  OT i2t { in(12), in(13), in(14), in(15), in(8),  in(9),
           in(4),  in(5),  in(6),  in(7),  in(10), in(11) };
  const float *p0w = in(16), *p0b = in(17), *p0g = in(18), *p0be = in(19);
  const float *p1w = in(20), *p1b = in(21), *p1g = in(22), *p1be = in(23);
  OT t2i { in(32), in(33), in(34), in(35), in(28), in(29),
           in(24), in(25), in(26), in(27), in(30), in(31) };
  const float *ent_text = in(36), *men_text = in(37);
  const float *ent_img  = in(38), *men_img  = in(39);

  // workspace arena
  float* W = (float*)d_ws;
  float* big1 = W;
  float* big2 = big1 + BIGSZ;
  float* big3 = big2 + BIGSZ;
  float* s1   = big3 + BIGSZ;
  float* s2   = s1 + SMALLSZ;
  float* s3   = s2 + SMALLSZ;
  float* s4   = s3 + SMALLSZ;
  float* pb   = s4 + SMALLSZ;
  float* ent0 = pb + PBSZ;
  float* ent1 = ent0 + VECSZ;
  float* men0 = ent1 + VECSZ;
  float* men1 = men0 + VECSZ;

  run_side(stream, ent_text, ent_img, i2t, t2i, ftW, ftB, fiW, fiB,
           p0w, p0b, p0g, p0be, p1w, p1b, p1g, p1be,
           big1, big2, big3, s1, s2, s3, s4, pb, ent0, ent1);
  run_side(stream, men_text, men_img, i2t, t2i, ftW, ftB, fiW, fiB,
           p0w, p0b, p0g, p0be, p1w, p1b, p1g, p1be,
           big1, big2, big3, s1, s2, s3, s4, pb, men0, men1);

  score_kernel<<<dim3(BB / 16, BB / 16), dim3(16, 16), 0, stream>>>(
      men0, ent0, men1, ent1, (float*)d_out);
}
